// AscendTriangleMixAttention_82179904242117
// MI455X (gfx1250) — matrix operand order repaired and checked by reading, not compiled, not run on hardware
//
#include <hip/hip_runtime.h>
#include <hip/hip_bf16.h>

// CDNA5 / gfx1250 flash-attention (causal), wave32, WMMA f16->f32.
// B=1, S=4096, H=16, D=128. Scores computed transposed (S^T = K*Q^T) so that
// the softmax'd P tile is already in WMMA A-fragment layout for P*V.
// v2: wave-uniform mask fast-path (only diagonal chunk pays mask VALU),
//     exp2-domain softmax (log2e folded into Q scale).

typedef __attribute__((ext_vector_type(16))) _Float16 v16h;
typedef __attribute__((ext_vector_type(8)))  float    v8f;

#define S_LEN 4096
#define NHEAD 16
#define DHEAD 128
#define BLK_Q 128              // queries per workgroup (8 waves x 16 rows)
#define CHUNK 32               // keys per iteration

// softmax in log2 domain: scores are (q.k)/sqrt(D) * log2(e)
#define QSCALE (0.08838834764831845f * 1.4426950408889634f)
#define NEGV   (-1.4426950408889634e9f)   // -1e9 * log2(e)

#if __has_builtin(__builtin_amdgcn_exp2f)
#define EXP2(x) __builtin_amdgcn_exp2f(x)
#else
#define EXP2(x) __expf((x) * 0.6931471805599453f)
#endif

union H16 { v16h v; uint4 u[2]; _Float16 h[16]; };

__global__ __launch_bounds__(256)
void AscendTriangleMixAttention_82179904242117_kernel(
    const float* __restrict__ q_g,
    const float* __restrict__ k_g,
    const float* __restrict__ v_g,
    float* __restrict__ out)
{
    // LDS staging: K chunk row-major [32 keys][128 d] f16 (8KB),
    //              V chunk transposed [128 d][32 keys] f16 (8KB).
    __shared__ __align__(16) _Float16 Ksh[CHUNK * DHEAD];
    __shared__ __align__(16) _Float16 Vsh[DHEAD * CHUNK];

    const int h      = blockIdx.y;                 // head
    const int q0blk  = blockIdx.x * BLK_Q;         // first query of block
    const int tid    = threadIdx.x;
    const int wave   = tid >> 5;
    const int lane   = tid & 31;
    const int lp     = lane & 15;                  // query column within tile
    const int half   = lane >> 4;                  // lane half (K stripe select)

    const int q0w    = q0blk + wave * 16;          // this wave's first query row
    const int qhi    = q0w + 15;
    const int qg     = q0w + lp;                   // this lane's query index

    // ---- Preload Q as 4 B-fragments (32 d-values each), pre-scaled, f16 ----
    // B-frag (32x16): lane = query col (lp); element e<8: d = 32c + 8*half + e, e>=8: d = 32c + 16 + 8*half + (e-8).
    v16h qb[4];
    {
        const float* qrow = q_g + ((size_t)qg * NHEAD + h) * DHEAD;
        #pragma unroll
        for (int c = 0; c < 4; ++c) {
            const float* p = qrow + 32 * c + 8 * half;
            H16 t;
            #pragma unroll
            for (int e = 0; e < 16; ++e)
                t.h[e] = (_Float16)(p[e + (e >= 8 ? 8 : 0)] * QSCALE);
            qb[c] = t.v;
        }
    }

    // ---- Output accumulators: 8 tiles of 16(q) x 16(d), C-layout ----
    v8f ofrag[8];
    #pragma unroll
    for (int n = 0; n < 8; ++n) ofrag[n] = (v8f){0.f,0.f,0.f,0.f,0.f,0.f,0.f,0.f};
    float m_run = -1e30f;   // running max (log2 domain; per query = lp, dup lane halves)
    float l_run = 0.0f;     // running sum

    // Cooperative-load mapping: 256 threads, 32 keys x (8 threads * 16 d).
    const int kk    = tid >> 3;          // key row within chunk
    const int dbase = (tid & 7) * 16;    // 16 consecutive d per thread

    const int jend = q0blk + BLK_Q;      // causal bound for whole block

    for (int j0 = 0; j0 < jend; j0 += CHUNK) {
        __syncthreads();   // previous iteration's LDS reads done

        // ---- Stage K chunk (row-major f16) and V chunk (transposed f16) ----
        {
            const size_t grow = ((size_t)(j0 + kk) * NHEAD + h) * DHEAD + dbase;
            const float4* kp = (const float4*)(k_g + grow);
            const float4* vp = (const float4*)(v_g + grow);
            float kf[16], vf[16];
            #pragma unroll
            for (int i = 0; i < 4; ++i) {
                float4 a = kp[i];
                kf[4*i+0]=a.x; kf[4*i+1]=a.y; kf[4*i+2]=a.z; kf[4*i+3]=a.w;
                float4 b = vp[i];
                vf[4*i+0]=b.x; vf[4*i+1]=b.y; vf[4*i+2]=b.z; vf[4*i+3]=b.w;
            }
            H16 t;
            #pragma unroll
            for (int i = 0; i < 16; ++i) t.h[i] = (_Float16)kf[i];
            uint4* kd = (uint4*)&Ksh[kk * DHEAD + dbase];
            kd[0] = t.u[0]; kd[1] = t.u[1];
            #pragma unroll
            for (int i = 0; i < 16; ++i)
                Vsh[(dbase + i) * CHUNK + kk] = (_Float16)vf[i];

            if (j0 + CHUNK < jend) {   // prefetch next chunk -> global_prefetch
                const size_t nrow = ((size_t)(j0 + CHUNK + kk) * NHEAD + h) * DHEAD + dbase;
                __builtin_prefetch(k_g + nrow, 0, 1);
                __builtin_prefetch(v_g + nrow, 0, 1);
            }
        }
        __syncthreads();   // chunk visible to all waves

        if (j0 > qhi) continue;   // wave-uniform: EXEC stays all-ones below

        // ---- S^T = K * Q^T : two 16(key) x 16(query) tiles ----
        v8f s0 = (v8f){0.f,0.f,0.f,0.f,0.f,0.f,0.f,0.f};
        v8f s1 = s0;
        #pragma unroll
        for (int c = 0; c < 4; ++c) {
            // A-frag (16 keys x 32 d): lane row = key, seg0 d = 32c+8*half,
            // seg1 = seg0+16.
            const int base0 = (0 * 16 + lp) * DHEAD + 32 * c + 8 * half;
            const int base1 = (1 * 16 + lp) * DHEAD + 32 * c + 8 * half;
            H16 a0, a1;
            a0.u[0] = *(const uint4*)&Ksh[base0];
            a0.u[1] = *(const uint4*)&Ksh[base0 + 16];
            a1.u[0] = *(const uint4*)&Ksh[base1];
            a1.u[1] = *(const uint4*)&Ksh[base1 + 16];
            s0 = __builtin_amdgcn_wmma_f32_16x16x32_f16(false, a0.v, false, qb[c],
                                                        (short)0, s0, false, false);
            s1 = __builtin_amdgcn_wmma_f32_16x16x32_f16(false, a1.v, false, qb[c],
                                                        (short)0, s1, false, false);
        }

        // ---- causal mask (diagonal chunk only) + online softmax ----
        // Wave-uniform: chunk is fully valid when last key <= wave's first query.
        float cmax = -1e30f;
        if (j0 + CHUNK - 1 <= q0w) {
            #pragma unroll
            for (int r = 0; r < 8; ++r) cmax = fmaxf(cmax, fmaxf(s0[r], s1[r]));
        } else {
            #pragma unroll
            for (int r = 0; r < 8; ++r) {
                int j = j0 + 8 * half + r;
                if (j > qg) s0[r] += NEGV;
                cmax = fmaxf(cmax, s0[r]);
            }
            #pragma unroll
            for (int r = 0; r < 8; ++r) {
                int j = j0 + 16 + 8 * half + r;
                if (j > qg) s1[r] += NEGV;
                cmax = fmaxf(cmax, s1[r]);
            }
        }
        cmax = fmaxf(cmax, __shfl_xor(cmax, 16, 32));
        const float m_new = fmaxf(m_run, cmax);
        const float alpha = EXP2(m_run - m_new);

        float lsum = 0.0f;
        H16 ap;   // P as WMMA A-fragment (16 q x 32 keys) -- layout matches!
        #pragma unroll
        for (int r = 0; r < 8; ++r) {
            float p = EXP2(s0[r] - m_new);
            lsum += p; ap.h[r] = (_Float16)p;
        }
        #pragma unroll
        for (int r = 0; r < 8; ++r) {
            float p = EXP2(s1[r] - m_new);
            lsum += p; ap.h[8 + r] = (_Float16)p;
        }
        lsum += __shfl_xor(lsum, 16, 32);
        l_run = l_run * alpha + lsum;
        m_run = m_new;

        // broadcast per-row alpha to the O accumulators (rows = 8*half + r)
        float arow[8];
        #pragma unroll
        for (int r = 0; r < 8; ++r) arow[r] = __shfl(alpha, 8 * half + r, 32);
        #pragma unroll
        for (int n = 0; n < 8; ++n)
            #pragma unroll
            for (int r = 0; r < 8; ++r) ofrag[n][r] *= arow[r];

        // ---- O += P * V  (8 d-tiles) ----
        #pragma unroll
        for (int n = 0; n < 8; ++n) {
            // B-frag (32 keys x 16 d): lane col = d = 16n+lp,
            // element e<8: key = 8*half+e, e>=8: key = 16+8*half+(e-8); Vsh is [d][key].
            const int vb = (16 * n + lp) * CHUNK + 8 * half;
            H16 b;
            b.u[0] = *(const uint4*)&Vsh[vb];
            b.u[1] = *(const uint4*)&Vsh[vb + 16];
            ofrag[n] = __builtin_amdgcn_wmma_f32_16x16x32_f16(false, ap.v, false, b.v,
                                                              (short)0, ofrag[n],
                                                              false, false);
        }
    }

    // ---- normalize and store ----
    float lrow[8];
    #pragma unroll
    for (int r = 0; r < 8; ++r)
        lrow[r] = 1.0f / __shfl(l_run, 8 * half + r, 32);

    #pragma unroll
    for (int n = 0; n < 8; ++n) {
        const int d = 16 * n + lp;
        #pragma unroll
        for (int r = 0; r < 8; ++r) {
            const int qrow = q0w + 8 * half + r;
            out[((size_t)qrow * NHEAD + h) * DHEAD + d] = ofrag[n][r] * lrow[r];
        }
    }
}

extern "C" void kernel_launch(void* const* d_in, const int* in_sizes, int n_in,
                              void* d_out, int out_size, void* d_ws, size_t ws_size,
                              hipStream_t stream) {
    const float* q = (const float*)d_in[0];
    const float* k = (const float*)d_in[1];
    const float* v = (const float*)d_in[2];
    // d_in[3] = attn_mask: known causal -1e9 triangle; applied analytically.
    float* out = (float*)d_out;

    dim3 grid(S_LEN / BLK_Q, NHEAD, 1);   // 32 x 16
    dim3 block(256, 1, 1);                 // 8 waves (wave32)
    AscendTriangleMixAttention_82179904242117_kernel<<<grid, block, 0, stream>>>(
        q, k, v, out);
}
